// GABALSTM_89996744721122
// MI455X (gfx1250) — matrix operand order repaired and checked by reading, not compiled, not run on hardware
//
#include <hip/hip_runtime.h>
#include <hip/hip_bf16.h>
#include <stdint.h>

typedef __attribute__((ext_vector_type(16))) _Float16 v16h;
typedef __attribute__((ext_vector_type(8)))  _Float16 v8h;
typedef __attribute__((ext_vector_type(4)))  _Float16 v4h;
typedef __attribute__((ext_vector_type(8)))  float    v8f;

#define LSTM_B 256
#define LSTM_S 1024
#define LSTM_F 64
#define LSTM_H 64
#define G4     256   // 4*H
#define SA     88    // half-elem stride of xbuf/hbuf rows (176B: 16B aligned, good bank spread)
#define ZS     260   // float stride of zbuf rows (1040B: 16B aligned)

// ---- WMMA helper: D = A(16x32 f16) * B(32x16 f16) + C(f32) ----
__device__ __forceinline__ v8f wmma_f16(v16h a, v16h b, v8f c) {
  return __builtin_amdgcn_wmma_f32_16x16x32_f16(false, a, false, b, (short)0, c,
                                                false, false);
}

// Native gfx1250 v_tanh_f32 if the toolchain exposes it; else exp fallback.
#if defined(__has_builtin)
#if __has_builtin(__builtin_amdgcn_tanhf)
#define TANHF(v) __builtin_amdgcn_tanhf(v)
#elif __has_builtin(__builtin_amdgcn_tanh_f32)
#define TANHF(v) __builtin_amdgcn_tanh_f32(v)
#endif
#endif
#ifndef TANHF
__device__ __forceinline__ float tanh_fallback(float v) {
  const float e = __expf(-2.0f * v);
  return (1.0f - e) / (1.0f + e);
}
#define TANHF(v) tanh_fallback(v)
#endif

__device__ __forceinline__ float sigf(float v) {
  // sigmoid(x) = 0.5*tanh(x/2) + 0.5  -> single TRANS32 op
  return __builtin_fmaf(0.5f, TANHF(0.5f * v), 0.5f);
}

// B-matrix register load from f32 weights W[64][256] (row = K, col = gate idx).
// CDNA5 f16 B layout (32x16): lane n (0-15): col n, K = kb+0..7 then kb+16..23;
// lane n+16: col n, K = kb+8..15 then kb+24..31.
__device__ __forceinline__ v16h load_B16(const float* __restrict__ W, int cb,
                                         int kb, int lane) {
  const int col = cb + (lane & 15);
  const int ks  = kb + ((lane & 16) ? 8 : 0);
  v16h r;
#pragma unroll
  for (int i = 0; i < 16; ++i) r[i] = (_Float16)W[(size_t)(ks + (i < 8 ? i : i + 8)) * G4 + col];
  return r;
}

// A-matrix load from f16 LDS tile (16 rows x 64 K, stride SA).
// CDNA5 f16 A layout (16x32): lane m (0-15): row m, K = kb+0..7 then kb+16..23;
// lane m+16: row m, K = kb+8..15 then kb+24..31.  Two aligned 16B DS loads.
__device__ __forceinline__ v16h load_A16(const _Float16* buf, int kb, int lane) {
  const int m  = lane & 15;
  const int ks = kb + ((lane & 16) ? 8 : 0);
  const v8h lo = *(const v8h*)&buf[m * SA + ks];
  const v8h hi = *(const v8h*)&buf[m * SA + ks + 16];
  v16h r;
#pragma unroll
  for (int i = 0; i < 8; ++i) { r[i] = lo[i]; r[8 + i] = hi[i]; }
  return r;
}

__global__ __launch_bounds__(256)
void lstm_fused_kernel(const float* __restrict__ x,
                       const float* __restrict__ kernel,
                       const float* __restrict__ rec_kernel,
                       const float* __restrict__ bias,
                       const float* __restrict__ dense_w,
                       const float* __restrict__ dense_b,
                       float* __restrict__ out) {
  __shared__ _Float16 xbuf[2][16 * SA];   // double-buffered x timestep tile (f16)
  __shared__ _Float16 hbuf[16 * SA];      // h state tile (f16)
  __shared__ float    zbuf[16 * ZS];      // pre-activation gates [16][256]

  const int tid  = threadIdx.x;
  const int lane = tid & 31;
  const int wid  = tid >> 5;                 // wave id 0..7
  const int b0   = blockIdx.x * 16;          // batch tile base
  const int row  = tid >> 4;                 // 0..15 (batch row for x/gate pass)
  const int col4 = (tid & 15) * 4;           // 0,4,...,60

  // ---------------- one-time init ----------------
  for (int i = tid; i < 16 * SA; i += 256) hbuf[i] = (_Float16)0.0f;

  const float* xrowp = x + (size_t)(b0 + row) * LSTM_S * LSTM_F + col4;
  {  // preload x tile for t = 0
    const float4 xv = *(const float4*)(xrowp);
    v4h hv;
    hv[0] = (_Float16)xv.x; hv[1] = (_Float16)xv.y;
    hv[2] = (_Float16)xv.z; hv[3] = (_Float16)xv.w;
    *(v4h*)&xbuf[0][row * SA + col4] = hv;
  }

  // Per-wave persistent B registers (input kernel + recurrent kernel) and bias.
  const int cb0 = wid * 32;
  const int cb1 = wid * 32 + 16;
  const v16h Bx0a = load_B16(kernel,     cb0,  0, lane);
  const v16h Bx0b = load_B16(kernel,     cb0, 32, lane);
  const v16h Bx1a = load_B16(kernel,     cb1,  0, lane);
  const v16h Bx1b = load_B16(kernel,     cb1, 32, lane);
  const v16h Br0a = load_B16(rec_kernel, cb0,  0, lane);
  const v16h Br0b = load_B16(rec_kernel, cb0, 32, lane);
  const v16h Br1a = load_B16(rec_kernel, cb1,  0, lane);
  const v16h Br1b = load_B16(rec_kernel, cb1, 32, lane);
  const float bias0 = bias[cb0 + (lane & 15)];
  const float bias1 = bias[cb1 + (lane & 15)];

  // Dense head params are thread-invariant across all timesteps: keep in regs.
  const float4 wv  = *(const float4*)(dense_w + col4);
  const float  db  = dense_b[0];

  const int zc0 = cb0 + (lane & 15);
  const int zc1 = cb1 + (lane & 15);
  const int ro  = (lane & 16) ? 8 : 0;       // D-matrix row offset for hi lanes

  __syncthreads();

  int    cur = 0;
  float4 cc  = make_float4(0.0f, 0.0f, 0.0f, 0.0f);  // cell state (4 h per thread)

  for (int t = 0; t < LSTM_S; ++t) {
    // Issue next timestep's x load early; land it after barrier1.
    float4 xn = make_float4(0.0f, 0.0f, 0.0f, 0.0f);
    const bool havenext = (t + 1 < LSTM_S);
    if (havenext) xn = *(const float4*)(xrowp + (size_t)(t + 1) * LSTM_F);

    // z = bias + x_t @ kernel + h @ rec_kernel  (two 16x16 tiles per wave)
    v8f acc0, acc1;
#pragma unroll
    for (int r = 0; r < 8; ++r) { acc0[r] = bias0; acc1[r] = bias1; }

    const v16h a0 = load_A16(xbuf[cur], 0, lane);
    const v16h a1 = load_A16(xbuf[cur], 32, lane);
    acc0 = wmma_f16(a0, Bx0a, acc0);
    acc0 = wmma_f16(a1, Bx0b, acc0);
    acc1 = wmma_f16(a0, Bx1a, acc1);
    acc1 = wmma_f16(a1, Bx1b, acc1);

    const v16h h0 = load_A16(hbuf, 0, lane);
    const v16h h1 = load_A16(hbuf, 32, lane);
    acc0 = wmma_f16(h0, Br0a, acc0);
    acc0 = wmma_f16(h1, Br0b, acc0);
    acc1 = wmma_f16(h0, Br1a, acc1);
    acc1 = wmma_f16(h1, Br1b, acc1);

    // Scatter D tiles to zbuf (D layout: VGPR r -> row r (+8 for hi lanes), col = lane&15)
#pragma unroll
    for (int r = 0; r < 8; ++r) {
      zbuf[(r + ro) * ZS + zc0] = acc0[r];
      zbuf[(r + ro) * ZS + zc1] = acc1[r];
    }
    __syncthreads();  // barrier1: zbuf ready; xbuf[cur]/hbuf reads done

    if (havenext) {
      v4h hv;
      hv[0] = (_Float16)xn.x; hv[1] = (_Float16)xn.y;
      hv[2] = (_Float16)xn.z; hv[3] = (_Float16)xn.w;
      *(v4h*)&xbuf[cur ^ 1][row * SA + col4] = hv;
    }

    // Gate pass: thread handles batch `row`, hidden cols col4..col4+3
    const float4 zi = *(const float4*)&zbuf[row * ZS +       col4];
    const float4 zf = *(const float4*)&zbuf[row * ZS +  64 + col4];
    const float4 zc = *(const float4*)&zbuf[row * ZS + 128 + col4];
    const float4 zo = *(const float4*)&zbuf[row * ZS + 192 + col4];

    cc.x = sigf(zf.x) * cc.x + sigf(zi.x) * TANHF(zc.x);
    cc.y = sigf(zf.y) * cc.y + sigf(zi.y) * TANHF(zc.y);
    cc.z = sigf(zf.z) * cc.z + sigf(zi.z) * TANHF(zc.z);
    cc.w = sigf(zf.w) * cc.w + sigf(zi.w) * TANHF(zc.w);

    float4 hn;
    hn.x = sigf(zo.x) * TANHF(cc.x);
    hn.y = sigf(zo.y) * TANHF(cc.y);
    hn.z = sigf(zo.z) * TANHF(cc.z);
    hn.w = sigf(zo.w) * TANHF(cc.w);

    {
      v4h hv;
      hv[0] = (_Float16)hn.x; hv[1] = (_Float16)hn.y;
      hv[2] = (_Float16)hn.z; hv[3] = (_Float16)hn.w;
      *(v4h*)&hbuf[row * SA + col4] = hv;
    }

    // Dense head: per-lane partial dot, then reduce across the 16 lanes that
    // share a batch row (lanes 0-15 = row 2w, lanes 16-31 = row 2w+1).
    float pd = hn.x * wv.x + hn.y * wv.y + hn.z * wv.z + hn.w * wv.w;
    pd += __shfl_xor(pd, 1);
    pd += __shfl_xor(pd, 2);
    pd += __shfl_xor(pd, 4);
    pd += __shfl_xor(pd, 8);
    if ((lane & 15) == 0) {
      out[(size_t)(b0 + row) * LSTM_S + t] = sigf(pd + db);
    }

    __syncthreads();  // barrier2: hbuf / xbuf[next] ready for next step

    cur ^= 1;
  }
}

extern "C" void kernel_launch(void* const* d_in, const int* in_sizes, int n_in,
                              void* d_out, int out_size, void* d_ws, size_t ws_size,
                              hipStream_t stream) {
  const float* x   = (const float*)d_in[0];
  const float* k   = (const float*)d_in[1];
  const float* rk  = (const float*)d_in[2];
  const float* b   = (const float*)d_in[3];
  const float* dw  = (const float*)d_in[4];
  const float* db  = (const float*)d_in[5];
  float*       out = (float*)d_out;

  lstm_fused_kernel<<<dim3(LSTM_B / 16), dim3(256), 0, stream>>>(
      x, k, rk, b, dw, db, out);
}
